// FieldCollapse_63651415327273
// MI455X (gfx1250) — matrix operand order repaired and checked by reading, not compiled, not run on hardware
//
#include <hip/hip_runtime.h>
#include <stdint.h>

// Problem constants (from reference setup_inputs)
#define B_   2
#define S_   2048
#define D_   1024
#define V_   32000
#define M_   (B_*S_)        // 4096 rows = B*S
#define WIN_ 2
#define THRESH_ 0.91f
#define EPS_    1e-8f

// GEMM tiling: block tile 64(M) x 256(N) x 32(K), 8 waves (2x4), wave tile 32x64
#define BM 64
#define BN 256
#define BK 32
#define NKSTEPS (D_/BK)     // 32

typedef __attribute__((ext_vector_type(16))) __bf16          bf16x16;
typedef __attribute__((ext_vector_type(8)))  float           f32x8;
typedef __attribute__((ext_vector_type(8)))  unsigned short  u16x8;
typedef __attribute__((ext_vector_type(4)))  int             i32x4;

union FragU { bf16x16 v; u16x8 h[2]; };

// ---- CDNA5 async global->LDS staging (ASYNCcnt), guarded with fallback ----
#if defined(__has_builtin)
#  if __has_builtin(__builtin_amdgcn_global_load_async_to_lds_b128)
#    define USE_ASYNC 1
#  endif
#endif
#ifndef USE_ASYNC
#  define USE_ASYNC 0
#endif

#if USE_ASYNC
// builtin signature (from compiler diagnostics): (AS1 i32x4*, AS3 i32x4*, imm offset, imm cpol)
#define ASYNC_B128(gp, lp)                                                  \
    __builtin_amdgcn_global_load_async_to_lds_b128(                         \
        (__attribute__((address_space(1))) i32x4*)(gp),                     \
        (__attribute__((address_space(3))) i32x4*)(lp), 0, 0)
#  if __has_builtin(__builtin_amdgcn_s_wait_asynccnt)
#    define WAIT_ASYNC(n) __builtin_amdgcn_s_wait_asynccnt(n)
#  else
#    define WAIT_ASYNC(n) asm volatile("s_wait_asynccnt %0" :: "i"(n) : "memory")
#  endif
#endif

__device__ __forceinline__ unsigned short f2bf(float f) {
    unsigned int u = __float_as_uint(f);
    u += 0x7FFFu + ((u >> 16) & 1u);          // round-to-nearest-even
    return (unsigned short)(u >> 16);
}

// ---------------------------------------------------------------------------
// fp32 -> bf16 conversion (bandwidth-trivial), 8 elements/thread
// ---------------------------------------------------------------------------
__global__ __launch_bounds__(256) void cvt_bf16_kernel(const float* __restrict__ in,
                                                       unsigned short* __restrict__ out,
                                                       int n) {
    int i = (blockIdx.x * 256 + threadIdx.x) * 8;
    if (i + 7 >= n) return;
    float4 a = *(const float4*)(in + i);
    float4 b = *(const float4*)(in + i + 4);
    u16x8 o;
    o[0] = f2bf(a.x); o[1] = f2bf(a.y); o[2] = f2bf(a.z); o[3] = f2bf(a.w);
    o[4] = f2bf(b.x); o[5] = f2bf(b.y); o[6] = f2bf(b.z); o[7] = f2bf(b.w);
    *(u16x8*)(out + i) = o;
}

// ---------------------------------------------------------------------------
// Coherence: one wave32 per (b,s). Per-lane partial 5x5 Gram over a strided
// slice of D, wave reduction via shfl_xor, Jacobi eigensolve on lane 0.
// ---------------------------------------------------------------------------
__global__ __launch_bounds__(256) void coherence_kernel(const float* __restrict__ field,
                                                        float* __restrict__ coh) {
    const int lane = threadIdx.x & 31;
    const int m    = blockIdx.x * 8 + (threadIdx.x >> 5);   // 8 waves / block
    const int b    = m >> 11;                               // S_ == 2048
    const int s    = m & (S_ - 1);

    bool valid[5];
    int  cnt = 0;
#pragma unroll
    for (int o = 0; o < 5; ++o) {
        int idx = s + o - WIN_;
        valid[o] = (idx >= 0) && (idx < S_);
        cnt += valid[o] ? 1 : 0;
    }
    const float icnt = 1.0f / (float)cnt;
    const float* base = field + (size_t)b * S_ * D_;

    float acc[15];
#pragma unroll
    for (int t = 0; t < 15; ++t) acc[t] = 0.f;

    for (int d = lane; d < D_; d += 32) {
        float w[5];
#pragma unroll
        for (int o = 0; o < 5; ++o) {
            int idx = s + o - WIN_;
            w[o] = valid[o] ? base[(size_t)idx * D_ + d] : 0.f;
        }
        float mean = (w[0] + w[1] + w[2] + w[3] + w[4]) * icnt;
        float c[5];
#pragma unroll
        for (int o = 0; o < 5; ++o) c[o] = valid[o] ? (w[o] - mean) : 0.f;
        int t = 0;
#pragma unroll
        for (int p = 0; p < 5; ++p)
#pragma unroll
            for (int q = p; q < 5; ++q) acc[t++] += c[p] * c[q];
    }

#pragma unroll
    for (int t = 0; t < 15; ++t)
#pragma unroll
        for (int off = 16; off > 0; off >>= 1)
            acc[t] += __shfl_xor(acc[t], off, 32);

    if (lane == 0) {
        float Am[5][5];
        int t = 0;
        for (int p = 0; p < 5; ++p)
            for (int q = p; q < 5; ++q) {
                float v = acc[t++] * icnt;
                Am[p][q] = v; Am[q][p] = v;
            }
        float trace = Am[0][0] + Am[1][1] + Am[2][2] + Am[3][3] + Am[4][4];
        for (int sweep = 0; sweep < 8; ++sweep) {
            for (int p = 0; p < 4; ++p)
                for (int q = p + 1; q < 5; ++q) {
                    float apq = Am[p][q];
                    if (fabsf(apq) > 1e-12f) {
                        float tau = (Am[q][q] - Am[p][p]) / (2.0f * apq);
                        float tt  = (tau >= 0.f ? 1.0f : -1.0f) /
                                    (fabsf(tau) + sqrtf(1.0f + tau * tau));
                        float cc  = rsqrtf(1.0f + tt * tt);
                        float ss  = tt * cc;
                        for (int k = 0; k < 5; ++k) {
                            float akp = Am[k][p], akq = Am[k][q];
                            Am[k][p] = cc * akp - ss * akq;
                            Am[k][q] = ss * akp + cc * akq;
                        }
                        for (int k = 0; k < 5; ++k) {
                            float apk = Am[p][k], aqk = Am[q][k];
                            Am[p][k] = cc * apk - ss * aqk;
                            Am[q][k] = ss * apk + cc * aqk;
                        }
                    }
                }
        }
        float lmax = Am[0][0];
        for (int k = 1; k < 5; ++k) lmax = fmaxf(lmax, Am[k][k]);
        coh[m] = lmax / (trace + EPS_);
    }
}

// ---------------------------------------------------------------------------
// GEMM: P[m,n] = (sum_k A[m,k]*W[n,k])^2 with bf16 WMMA, f32 accumulate.
// 8 waves (2x4), wave tile 32x64 = 2x4 WMMA fragments (A frags reused 4x).
// Staging via GLOBAL_LOAD_ASYNC_TO_LDS (ASYNCcnt) when available; W (65.5 MB
// bf16) is fully L2-resident on MI455X (192 MB L2).
// ---------------------------------------------------------------------------
__global__ __launch_bounds__(256) void gemm_p_kernel(const unsigned short* __restrict__ A,  // [M_,D_] bf16
                                                     const unsigned short* __restrict__ W,  // [V_,D_] bf16
                                                     float* __restrict__ P) {               // [M_,V_] proj^2
    __shared__ __attribute__((aligned(16))) unsigned short ldsA[2][BM * BK];  // 8 KB
    __shared__ __attribute__((aligned(16))) unsigned short ldsB[2][BN * BK];  // 32 KB

    const int tid  = threadIdx.x;
    const int lane = tid & 31;
    const int wv   = tid >> 5;
    const int wm   = wv >> 2;          // 0..1
    const int wn   = wv & 3;           // 0..3
    const int m0   = blockIdx.y * BM;
    const int n0   = blockIdx.x * BN;

    // staging: A tile 64x32 -> thread t: row t>>2, chunk t&3 (8 bf16 = 16 B)
    //          B tile 256x32 -> thread t: row t, 4 chunks of 8 bf16
    const int arow = tid >> 2, ach = tid & 3;
    const size_t aG = (size_t)(m0 + arow) * D_ + ach * 8;
    const size_t bG = (size_t)(n0 + tid) * D_;
    const int aL = arow * BK + ach * 8;
    const int bL = tid * BK;

#if USE_ASYNC
    ASYNC_B128(A + aG, &ldsA[0][aL]);
#pragma unroll
    for (int ch = 0; ch < 4; ++ch)
        ASYNC_B128(W + bG + ch * 8, &ldsB[0][bL + ch * 8]);
#else
    {
        u16x8 ar = *(const u16x8*)(A + aG);
        u16x8 br[4];
#pragma unroll
        for (int ch = 0; ch < 4; ++ch) br[ch] = *(const u16x8*)(W + bG + ch * 8);
        *(u16x8*)&ldsA[0][aL] = ar;
#pragma unroll
        for (int ch = 0; ch < 4; ++ch) *(u16x8*)&ldsB[0][bL + ch * 8] = br[ch];
    }
#endif

    const f32x8 zero8 = {0.f, 0.f, 0.f, 0.f, 0.f, 0.f, 0.f, 0.f};
    f32x8 acc[2][4];
#pragma unroll
    for (int i = 0; i < 2; ++i)
#pragma unroll
        for (int j = 0; j < 4; ++j) acc[i][j] = zero8;

    const int hsel = lane >> 4;        // wave32 WMMA lane-half select
    const int l15  = lane & 15;

#if !USE_ASYNC
    u16x8 ar_n, br_n[4];
#endif

    for (int kt = 0; kt < NKSTEPS; ++kt) {
        const int cur = kt & 1;
        const int nxt = cur ^ 1;
        const size_t kon = (size_t)(kt + 1) * BK;

#if USE_ASYNC
        if (kt + 1 < NKSTEPS) {
            ASYNC_B128(A + aG + kon, &ldsA[nxt][aL]);
#pragma unroll
            for (int ch = 0; ch < 4; ++ch)
                ASYNC_B128(W + bG + kon + ch * 8, &ldsB[nxt][bL + ch * 8]);
        }
        if (kt + 2 < NKSTEPS) {
            __builtin_prefetch(A + aG + (size_t)(kt + 2) * BK, 0, 1);
            __builtin_prefetch(W + bG + (size_t)(kt + 2) * BK, 0, 1);
        }
        // async loads complete in order: cnt<=5 => tile kt's 5 ops landed
        if (kt + 1 < NKSTEPS) { WAIT_ASYNC(5); } else { WAIT_ASYNC(0); }
        __syncthreads();
#else
        __syncthreads();
        if (kt + 1 < NKSTEPS) {
            ar_n = *(const u16x8*)(A + aG + kon);
#pragma unroll
            for (int ch = 0; ch < 4; ++ch) br_n[ch] = *(const u16x8*)(W + bG + kon + ch * 8);
        }
        if (kt + 2 < NKSTEPS) {
            __builtin_prefetch(A + aG + (size_t)(kt + 2) * BK, 0, 1);
            __builtin_prefetch(W + bG + (size_t)(kt + 2) * BK, 0, 1);
        }
#endif

        // A 16x32 bf16 frags: lanes 0-15 rows, K {0..7,16..23}; lanes 16-31 K {8..15,24..31}
        FragU a0, a1;
        {
            int rb = (wm * 32 + l15) * BK;
            a0.h[0] = *(const u16x8*)&ldsA[cur][rb + hsel * 8];
            a0.h[1] = *(const u16x8*)&ldsA[cur][rb + 16 + hsel * 8];
            rb = (wm * 32 + 16 + l15) * BK;
            a1.h[0] = *(const u16x8*)&ldsA[cur][rb + hsel * 8];
            a1.h[1] = *(const u16x8*)&ldsA[cur][rb + 16 + hsel * 8];
        }
#pragma unroll
        for (int j = 0; j < 4; ++j) {
            // B 32x16 bf16 frag: lane col = l15; lanes 0-15 K {0..7,16..23}; lanes 16-31 K {8..15,24..31}
            FragU b;
            const int nb = (wn * 64 + j * 16 + l15) * BK + hsel * 8;
            b.h[0] = *(const u16x8*)&ldsB[cur][nb];
            b.h[1] = *(const u16x8*)&ldsB[cur][nb + 16];
            acc[0][j] = __builtin_amdgcn_wmma_f32_16x16x32_bf16(false, a0.v, false, b.v, (short)0, acc[0][j], false, false);
            acc[1][j] = __builtin_amdgcn_wmma_f32_16x16x32_bf16(false, a1.v, false, b.v, (short)0, acc[1][j], false, false);
        }
        __syncthreads();

#if !USE_ASYNC
        if (kt + 1 < NKSTEPS) {
            *(u16x8*)&ldsA[nxt][aL] = ar_n;
#pragma unroll
            for (int ch = 0; ch < 4; ++ch) *(u16x8*)&ldsB[nxt][bL + ch * 8] = br_n[ch];
        }
#endif
    }

    // C/D layout: VGPR v -> M = v (lanes 0-15) or v+8 (lanes 16-31); N = l15
    const int mBase = m0 + wm * 32;
    const int nBase = n0 + wn * 64;
#pragma unroll
    for (int i = 0; i < 2; ++i)
#pragma unroll
        for (int j = 0; j < 4; ++j)
#pragma unroll
            for (int v = 0; v < 8; ++v) {
                const int mm = mBase + i * 16 + v + 8 * hsel;
                const int nn = nBase + j * 16 + l15;
                float x = acc[i][j][v];
                P[(size_t)mm * V_ + nn] = x * x;
            }
}

// ---------------------------------------------------------------------------
// Row-wise normalize + Gumbel-max categorical sample + threshold gate.
// One 256-thread block per row; row (128 KB) stays L2-resident between passes.
// ---------------------------------------------------------------------------
__device__ __forceinline__ float u01hash(unsigned long long x) {
    x ^= 42ull * 0xD1342543DE82EF95ull;                  // fold in seed 42
    x += 0x9E3779B97F4A7C15ull;
    x = (x ^ (x >> 30)) * 0xBF58476D1CE4E5B9ull;
    x = (x ^ (x >> 27)) * 0x94D049BB133111EBull;
    x ^= x >> 31;
    return ((float)(unsigned int)(x >> 40) + 0.5f) * (1.0f / 16777216.0f);
}

__global__ __launch_bounds__(256) void normalize_sample_kernel(float* __restrict__ P,
                                                               const float* __restrict__ coh,
                                                               int* __restrict__ tokens) {
    const int m   = blockIdx.x;
    const int tid = threadIdx.x;
    float* row = P + (size_t)m * V_;
    __shared__ float sred[256];
    __shared__ int   sidx[256];

    float sum = 0.f;
    for (int v = tid; v < V_; v += 256) sum += row[v];
    sred[tid] = sum;
    __syncthreads();
    for (int off = 128; off > 0; off >>= 1) {
        if (tid < off) sred[tid] += sred[tid + off];
        __syncthreads();
    }
    const float inv = 1.0f / (sred[0] + EPS_);
    __syncthreads();

    float best = -3.4e38f;
    int   bi   = 0;
    for (int v = tid; v < V_; v += 256) {
        float pr = row[v] * inv;
        row[v] = pr;
        float lg = __logf(pr + 1e-30f);
        float u  = u01hash(((unsigned long long)m << 32) | (unsigned int)v);
        float g  = -__logf(-__logf(u));
        float sc = lg + g;
        if (sc > best) { best = sc; bi = v; }
    }
    sred[tid] = best; sidx[tid] = bi;
    __syncthreads();
    for (int off = 128; off > 0; off >>= 1) {
        if (tid < off && sred[tid + off] > sred[tid]) {
            sred[tid] = sred[tid + off];
            sidx[tid] = sidx[tid + off];
        }
        __syncthreads();
    }
    if (tid == 0) tokens[m] = (coh[m] > THRESH_) ? sidx[0] : -1;
}

// ---------------------------------------------------------------------------
extern "C" void kernel_launch(void* const* d_in, const int* in_sizes, int n_in,
                              void* d_out, int out_size, void* d_ws, size_t ws_size,
                              hipStream_t stream) {
    const float* field = (const float*)d_in[0];   // (B,S,D) fp32
    const float* mops  = (const float*)d_in[1];   // (V,D)   fp32

    int*   tokens = (int*)d_out;                  // first M_ elements (int32 bits)
    float* probs  = (float*)d_out + M_;           // then M_*V_ floats

    char* w = (char*)d_ws;
    unsigned short* Abf = (unsigned short*)w;                                   // M_*D_*2  bytes
    unsigned short* Wbf = (unsigned short*)(w + (size_t)M_ * D_ * 2);           // V_*D_*2  bytes
    float*          coh = (float*)(w + (size_t)M_ * D_ * 2 + (size_t)V_ * D_ * 2);

    { int n = M_ * D_; cvt_bf16_kernel<<<(n / 8 + 255) / 256, 256, 0, stream>>>(field, Abf, n); }
    { int n = V_ * D_; cvt_bf16_kernel<<<(n / 8 + 255) / 256, 256, 0, stream>>>(mops,  Wbf, n); }

    coherence_kernel<<<M_ / 8, 256, 0, stream>>>(field, coh);

    dim3 grid(V_ / BN, M_ / BM);   // 125 x 64, exact
    gemm_p_kernel<<<grid, 256, 0, stream>>>(Abf, Wbf, probs);

    normalize_sample_kernel<<<M_, 256, 0, stream>>>(probs, coh, tokens);
}
